// SequenceRouter_37520834297859
// MI455X (gfx1250) — matrix operand order repaired and checked by reading, not compiled, not run on hardware
//
#include <hip/hip_runtime.h>
#include <hip/hip_bf16.h>

typedef __bf16 bf16_t;
typedef __attribute__((ext_vector_type(16))) __bf16 v16bf;
typedef __attribute__((ext_vector_type(8)))  float  v8f;

// fragment loaded as two 16B vectors (32 contiguous bytes per lane)
union Frag16 { v16bf v; uint4 q[2]; };

// ---------------- problem dims ----------------
#define BATCH 8
#define TIN   2048
#define FIN   80
#define T2    1024      // after conv0 (stride 2)
#define F2    40
#define S1    512       // after conv1 (stride 2)
#define F1    20
#define NF    64
#define K1    576       // 3*3*64 im2col K for conv1
#define KT1   (K1/32)   // 18 k-tiles
#define M1    (BATCH*S1*F1)   // 81920 output pixels of conv1
#define KP    1280      // 20*64  proj K
#define KTP   (KP/32)   // 40 k-tiles
#define MP    (BATCH*S1)      // 4096 positions
#define NI    96        // 3*32 windowed input capsules

// ---------------- workspace layout (bytes) ----------------
#define OFF_X0    ((size_t)0)                       // bf16 [8,1024,40,64]
#define OFF_WCAT  ((size_t)41943040)                // bf16 packed [8 nt][18 kt][32 lane][16]
#define OFF_PROJW ((size_t)42090496)                // bf16 packed [2 nt][40 kt][32 lane][16]
#define OFF_X1    ((size_t)42172416)                // bf16 [81920,64] == [4096,1280]
#define OFF_EMB0  ((size_t)52658176)                // f32  [4096,32]
#define OFF_CAPS  ((size_t)53182464)                // f32  [4096,32,8]
#define OFF_V0    ((size_t)57376768)                // f32  [4096,32,8]

// =====================================================================
// Kernel 1: conv0 maxout (3x3 s2, 1->64) + feat_mask(ceil(len/2)) -> bf16
// =====================================================================
__global__ void conv0_kernel(const float* __restrict__ x,
                             const float* __restrict__ w1, const float* __restrict__ b1,
                             const float* __restrict__ w2, const float* __restrict__ b2,
                             const int* __restrict__ lens,
                             bf16_t* __restrict__ x0)
{
    const int idx = blockIdx.x * 256 + threadIdx.x;          // < 8*1024*40*64
    const int c  = idx & 63;
    const int f2 = (idx >> 6) % F2;
    const int t2 = (idx >> 6) / F2 % T2;
    const int b  = idx / (64 * F2 * T2);
    float v1 = b1[c], v2 = b2[c];
    #pragma unroll
    for (int kh = 0; kh < 3; ++kh) {
        const int it = 2 * t2 + kh;                           // pad_before = 0
        if (it >= TIN) continue;
        #pragma unroll
        for (int kw = 0; kw < 3; ++kw) {
            const int ifr = 2 * f2 + kw;
            if (ifr >= FIN) continue;
            const float xv = x[(b * TIN + it) * FIN + ifr];
            const int wi = (kh * 3 + kw) * 64 + c;
            v1 = fmaf(xv, w1[wi], v1);
            v2 = fmaf(xv, w2[wi], v2);
        }
    }
    float v = fmaxf(v1, v2);
    const int vl = (lens[b] + 1) >> 1;                        // ceil(len/2)
    if (t2 >= vl) v = 0.f;
    x0[idx] = (bf16_t)v;
}

// =====================================================================
// Kernel 2: repack weights into WMMA B-fragment order (bf16).
//   Fragment layout used here (contiguous K, matches A loads): lane = {khalf(1b), col(4b)}; element e holds
//   K = kt*32 + khalf*16 + e, N = nt*16 + col. Each lane owns 32 contig B.
// =====================================================================
__global__ void repack_kernel(const float* __restrict__ c1w1, const float* __restrict__ c1w2,
                              const float* __restrict__ projw,
                              bf16_t* __restrict__ wcat, bf16_t* __restrict__ pw)
{
    const int idx = blockIdx.x * 256 + threadIdx.x;           // < 73728 + 40960
    if (idx < 8 * KT1 * 32 * 16) {
        const int e    = idx & 15;
        const int lane = (idx >> 4) & 31;
        const int kt   = (idx >> 9) % KT1;
        const int nt   = idx / (16 * 32 * KT1);
        const int n = nt * 16 + (lane & 15);
        const int k = kt * 32 + (lane >> 4) * 16 + e;
        const float v = (n < 64) ? c1w1[k * 64 + n] : c1w2[k * 64 + (n - 64)];
        wcat[idx] = (bf16_t)v;
    } else {
        const int j = idx - 8 * KT1 * 32 * 16;
        if (j < 2 * KTP * 32 * 16) {
            const int e    = j & 15;
            const int lane = (j >> 4) & 31;
            const int kt   = (j >> 9) % KTP;
            const int nt   = j / (16 * 32 * KTP);
            const int n = nt * 16 + (lane & 15);
            const int k = kt * 32 + (lane >> 4) * 16 + e;
            pw[j] = (bf16_t)projw[k * 32 + n];
        }
    }
}

// =====================================================================
// Kernel 3: conv1 implicit GEMM via v_wmma_f32_16x16x32_bf16
//   M=81920 pixels, K=576, N=128 (maxout pair); epilogue maxout+bias+mask.
//   A staged as im2col tile in LDS (ds_load_b128 x2 per fragment);
//   B read as packed fragments (global_load_b128 x2 per fragment).
// =====================================================================
__global__ void conv1_wmma_kernel(const bf16_t* __restrict__ x0,
                                  const bf16_t* __restrict__ wcat,
                                  const float* __restrict__ b1, const float* __restrict__ b2,
                                  const int* __restrict__ lens,
                                  bf16_t* __restrict__ x1)
{
    __shared__ bf16_t Alds[16][K1 + 8];
    __shared__ float  Clds[16][128];

    const int t    = threadIdx.x;
    const int m0   = blockIdx.x * 16;

    // ---- stage im2col A tile [16 x 576] into LDS ----
    for (int idx = t; idx < 16 * K1; idx += 256) {
        const int r = idx / K1, k = idx % K1;
        const int m = m0 + r;
        const int b  = m / (S1 * F1);
        const int rr = m % (S1 * F1);
        const int tt = rr / F1, f = rr % F1;
        const int kh = k / (3 * NF);
        const int kw = (k / NF) % 3;
        const int ci = k & 63;
        const int it = 2 * tt + kh;                            // pad_before = 0
        const int ifr = 2 * f + kw;
        bf16_t v = (bf16_t)0.f;
        if (it < T2 && ifr < F2)
            v = x0[((b * T2 + it) * F2 + ifr) * 64 + ci];
        Alds[r][k] = v;
    }
    __syncthreads();

    const int wave = t >> 5;             // 8 waves -> 8 N-tiles of 16 (128 cols)
    const int lane = t & 31;
    const int row   = lane & 15;
    const int kbase = (lane < 16) ? 0 : 16;    // A fragment K base
    const int ncol  = wave * 16 + (lane & 15);
    const bf16_t* bbase = wcat + (((size_t)wave * KT1) * 32 + lane) * 16;

    v8f acc = {};
    for (int kt = 0; kt < KT1; ++kt) {
        if (kt + 1 < KT1)
            __builtin_prefetch((const void*)(bbase + (size_t)(kt + 1) * 512), 0, 0);
        Frag16 fa, fb;
        const bf16_t* arow = &Alds[row][kt * 32 + kbase];
        fa.q[0] = *(const uint4*)(arow);          // K = kbase .. kbase+7
        fa.q[1] = *(const uint4*)(arow + 8);      // K = kbase+8 .. kbase+15
        const uint4* bp = (const uint4*)(bbase + (size_t)kt * 512);
        fb.q[0] = bp[0];
        fb.q[1] = bp[1];
        acc = __builtin_amdgcn_wmma_f32_16x16x32_bf16(false, fa.v, false, fb.v,
                                                      (short)0, acc, false, false);
    }

    // D layout: vgpr r -> row (lane<16 ? r : r+8), col = lane&15
    #pragma unroll
    for (int r = 0; r < 8; ++r) {
        const int ml = (lane < 16) ? r : (r + 8);
        Clds[ml][ncol] = acc[r];
    }
    __syncthreads();

    // ---- epilogue: bias + maxout + mask(ceil(len/4)) -> bf16 ----
    for (int idx = t; idx < 16 * 64; idx += 256) {
        const int r = idx >> 6, c = idx & 63;
        const int m = m0 + r;
        const int b  = m / (S1 * F1);
        const int tt = (m % (S1 * F1)) / F1;
        const int vl = (lens[b] + 3) >> 2;
        float v = fmaxf(Clds[r][c] + b1[c], Clds[r][c + 64] + b2[c]);
        if (tt >= vl) v = 0.f;
        x1[(size_t)m * 64 + c] = (bf16_t)v;
    }
}

// =====================================================================
// Kernel 4: projection GEMM [4096,1280]x[1280,32] via WMMA bf16, +bias
//   A fragments: two aligned uint4 loads per lane straight from x1.
// =====================================================================
__global__ void proj_wmma_kernel(const bf16_t* __restrict__ x1,
                                 const bf16_t* __restrict__ pw,
                                 const float* __restrict__ pb,
                                 float* __restrict__ emb0)
{
    const int wave = threadIdx.x >> 5;              // 4 waves / block
    const int lane = threadIdx.x & 31;
    const int task = blockIdx.x * 4 + wave;         // 512 tasks: 256 m-tiles x 2 n-tiles
    const int mt = task >> 1, nt = task & 1;

    const int row   = mt * 16 + (lane & 15);
    const int kbase = (lane < 16) ? 0 : 16;
    const int ncol  = nt * 16 + (lane & 15);
    const bf16_t* bbase = pw + (((size_t)nt * KTP) * 32 + lane) * 16;
    const bf16_t* abase = x1 + (size_t)row * KP + kbase;

    v8f acc = {};
    for (int kt = 0; kt < KTP; ++kt) {
        Frag16 fa, fb;
        const bf16_t* arow = abase + kt * 32;
        fa.q[0] = *(const uint4*)(arow);          // K = kbase .. kbase+7
        fa.q[1] = *(const uint4*)(arow + 8);      // K = kbase+8 .. kbase+15
        const uint4* bp = (const uint4*)(bbase + (size_t)kt * 512);
        fb.q[0] = bp[0];
        fb.q[1] = bp[1];
        acc = __builtin_amdgcn_wmma_f32_16x16x32_bf16(false, fa.v, false, fb.v,
                                                      (short)0, acc, false, false);
    }
    const float bias = pb[ncol];
    #pragma unroll
    for (int r = 0; r < 8; ++r) {
        const int m = mt * 16 + ((lane < 16) ? r : (r + 8));
        emb0[m * 32 + ncol] = acc[r] + bias;
    }
}

// =====================================================================
// Kernel 5: encoder conv (3x3 s1, 1->8) maxout + mask + squash + LN(256)
//   one block per (b,s); thread t -> (n = t/8, d = t%8)
// =====================================================================
__global__ void enc_kernel(const float* __restrict__ emb0,
                           const float* __restrict__ w1, const float* __restrict__ b1,
                           const float* __restrict__ w2, const float* __restrict__ b2,
                           const int* __restrict__ lens,
                           float* __restrict__ capsOut)
{
    const int bs = blockIdx.x, b = bs >> 9, s = bs & 511;
    const int t = threadIdx.x, n = t >> 3, d = t & 7;

    float v1 = b1[d], v2 = b2[d];
    #pragma unroll
    for (int kh = 0; kh < 3; ++kh) {
        const int ss = s + kh - 1;
        if (ss < 0 || ss >= S1) continue;
        #pragma unroll
        for (int kw = 0; kw < 3; ++kw) {
            const int nn = n + kw - 1;
            if (nn < 0 || nn >= 32) continue;
            const float xv = emb0[((b << 9) + ss) * 32 + nn];
            const int wi = (kh * 3 + kw) * 8 + d;
            v1 = fmaf(xv, w1[wi], v1);
            v2 = fmaf(xv, w2[wi], v2);
        }
    }
    float v = fmaxf(v1, v2);
    const int vl = (lens[b] + 3) >> 2;
    if (s >= vl) v = 0.f;

    // squash over d (8 adjacent lanes)
    float sq = v * v;
    sq += __shfl_xor(sq, 1, 8); sq += __shfl_xor(sq, 2, 8); sq += __shfl_xor(sq, 4, 8);
    v = v * (sq / (1.f + sq)) * rsqrtf(sq + 1e-6f);

    // layernorm over 256
    __shared__ float red[16];
    float s1 = v, s2 = v * v;
    for (int off = 16; off; off >>= 1) {
        s1 += __shfl_down(s1, off, 32);
        s2 += __shfl_down(s2, off, 32);
    }
    if ((t & 31) == 0) { red[t >> 5] = s1; red[8 + (t >> 5)] = s2; }
    __syncthreads();
    if (t == 0) {
        float a = 0.f, bb = 0.f;
        for (int w = 0; w < 8; ++w) { a += red[w]; bb += red[8 + w]; }
        const float mean = a / 256.f;
        red[0] = mean;
        red[1] = rsqrtf(bb / 256.f - mean * mean + 1e-3f);
    }
    __syncthreads();
    capsOut[(bs * 32 + n) * 8 + d] = (v - red[0]) * red[1];
}

// =====================================================================
// Kernel 6: fused window_concat + u_hat + dynamic routing (3 iters)
//   one block per (b,s); u recomputed from the 3KB LDS window each use.
//   MODE 0: layernorm(v) -> v0 buffer.  MODE 1: caps lengths + LN -> out.
// =====================================================================
template <int O, bool MASK0, int MODE>
__global__ void caps_route_kernel(const float* __restrict__ embIn,
                                  const float* __restrict__ Wm,
                                  const float* __restrict__ Bm,
                                  float* __restrict__ vout,
                                  float* __restrict__ outFinal)
{
    const int bs = blockIdx.x, b = bs >> 9, s = bs & 511;
    const int t = threadIdx.x;

    __shared__ float win[NI][8];
    __shared__ float blog[NI][32];
    __shared__ float cc[NI][32];
    __shared__ float vsh[32][8];
    __shared__ float red[16];

    // window gather: i = j*32+n  ->  emb[b, s+j-1, n, :], zero padded
    for (int idx = t; idx < NI * 8; idx += 256) {
        const int i = idx >> 3, k = idx & 7;
        const int j = i >> 5, n = i & 31;
        const int ss = s + j - 1;
        win[i][k] = (ss >= 0 && ss < S1) ? embIn[(((b << 9) + ss) * 32 + n) * 8 + k] : 0.f;
    }
    for (int idx = t; idx < NI * 32; idx += 256) (&blog[0][0])[idx] = 0.f;
    __syncthreads();

    const int o = t >> 3, d = t & 7;

    for (int it = 0; it < 3; ++it) {
        // coupling coefficients: softmax over o per input capsule i
        if (t < NI) {
            const int i = t;
            float m = -1e30f;
            for (int oo = 0; oo < O; ++oo) {
                const float l = blog[i][oo] + ((MASK0 && oo == 0) ? -1e9f : 0.f);
                m = fmaxf(m, l);
            }
            float den = 0.f;
            for (int oo = 0; oo < O; ++oo) {
                const float l = blog[i][oo] + ((MASK0 && oo == 0) ? -1e9f : 0.f);
                const float e = __expf(l - m);
                cc[i][oo] = e; den += e;
            }
            const float inv = 1.f / den;
            for (int oo = 0; oo < O; ++oo) cc[i][oo] *= inv;
        }
        __syncthreads();

        // s[o,d] = sum_i c[i,o] * u[i,o,d];  u recomputed from window
        if (o < O) {
            float sacc = 0.f;
            for (int i = 0; i < NI; ++i) {
                float u = Bm[(i * O + o) * 8 + d];
                const float* wr = Wm + (size_t)((i * O + o) * 8 + d) * 8;
                #pragma unroll
                for (int k = 0; k < 8; ++k) u = fmaf(wr[k], win[i][k], u);
                sacc = fmaf(cc[i][o], u, sacc);
            }
            float sq = sacc * sacc;
            sq += __shfl_xor(sq, 1, 8); sq += __shfl_xor(sq, 2, 8); sq += __shfl_xor(sq, 4, 8);
            vsh[o][d] = sacc * (sq / (1.f + sq)) * rsqrtf(sq + 1e-6f);
        }
        __syncthreads();

        // agreement update (skipped after last iteration: vs[-1] is returned)
        if (it < 2) {
            for (int idx = t; idx < NI * O; idx += 256) {
                const int i = idx / O, oo = idx % O;
                float dot = 0.f;
                #pragma unroll
                for (int dd = 0; dd < 8; ++dd) {
                    float u = Bm[(i * O + oo) * 8 + dd];
                    const float* wr = Wm + (size_t)((i * O + oo) * 8 + dd) * 8;
                    for (int k = 0; k < 8; ++k) u = fmaf(wr[k], win[i][k], u);
                    dot = fmaf(u, vsh[oo][dd], dot);
                }
                blog[i][oo] += dot;
            }
            __syncthreads();
        }
    }

    if (MODE == 0) {
        // layernorm over flattened [32*8]
        const float x = vsh[o][d];
        float s1 = x, s2 = x * x;
        for (int off = 16; off; off >>= 1) {
            s1 += __shfl_down(s1, off, 32);
            s2 += __shfl_down(s2, off, 32);
        }
        if ((t & 31) == 0) { red[t >> 5] = s1; red[8 + (t >> 5)] = s2; }
        __syncthreads();
        if (t == 0) {
            float a = 0.f, bb = 0.f;
            for (int w = 0; w < 8; ++w) { a += red[w]; bb += red[8 + w]; }
            const float mean = a / 256.f;
            red[0] = mean;
            red[1] = rsqrtf(bb / 256.f - mean * mean + 1e-3f);
        }
        __syncthreads();
        vout[(bs * 32 + o) * 8 + d] = (x - red[0]) * red[1];
    } else {
        // class-capsule lengths + layernorm over O
        if (t < O) {
            float sq = 1e-6f;
            #pragma unroll
            for (int dd = 0; dd < 8; ++dd) { const float v = vsh[t][dd]; sq = fmaf(v, v, sq); }
            cc[0][t] = sqrtf(sq);
        }
        __syncthreads();
        if (t == 0) {
            float a = 0.f;
            for (int oo = 0; oo < O; ++oo) a += cc[0][oo];
            const float mean = a / (float)O;
            float vv = 0.f;
            for (int oo = 0; oo < O; ++oo) { const float dl = cc[0][oo] - mean; vv += dl * dl; }
            red[0] = mean;
            red[1] = rsqrtf(vv / (float)O + 1e-3f);
        }
        __syncthreads();
        if (t < O) outFinal[bs * O + t] = (cc[0][t] - red[0]) * red[1];
    }
}

// =====================================================================
extern "C" void kernel_launch(void* const* d_in, const int* in_sizes, int n_in,
                              void* d_out, int out_size, void* d_ws, size_t ws_size,
                              hipStream_t stream)
{
    (void)in_sizes; (void)n_in; (void)out_size; (void)ws_size;

    const float* inputs  = (const float*)d_in[0];
    const float* c0w1    = (const float*)d_in[1];
    const float* c0b1    = (const float*)d_in[2];
    const float* c0w2    = (const float*)d_in[3];
    const float* c0b2    = (const float*)d_in[4];
    const float* c1w1    = (const float*)d_in[5];
    const float* c1b1    = (const float*)d_in[6];
    const float* c1w2    = (const float*)d_in[7];
    const float* c1b2    = (const float*)d_in[8];
    const float* projw   = (const float*)d_in[9];
    const float* projb   = (const float*)d_in[10];
    const float* ew1     = (const float*)d_in[11];
    const float* eb1     = (const float*)d_in[12];
    const float* ew2     = (const float*)d_in[13];
    const float* eb2     = (const float*)d_in[14];
    const float* W0      = (const float*)d_in[15];
    const float* B0      = (const float*)d_in[16];
    const float* W1      = (const float*)d_in[17];
    const float* B1      = (const float*)d_in[18];
    const int*   lens    = (const int*)d_in[19];
    float*       out     = (float*)d_out;

    char* ws = (char*)d_ws;
    bf16_t* x0    = (bf16_t*)(ws + OFF_X0);
    bf16_t* wcat  = (bf16_t*)(ws + OFF_WCAT);
    bf16_t* pwbf  = (bf16_t*)(ws + OFF_PROJW);
    bf16_t* x1    = (bf16_t*)(ws + OFF_X1);
    float*  emb0  = (float*)(ws + OFF_EMB0);
    float*  caps  = (float*)(ws + OFF_CAPS);
    float*  v0    = (float*)(ws + OFF_V0);

    // 1) conv0 maxout + mask -> bf16  (8*1024*40*64 = 20,971,520 elems)
    conv0_kernel<<<81920, 256, 0, stream>>>(inputs, c0w1, c0b1, c0w2, c0b2, lens, x0);

    // 2) weight repack into WMMA fragment order
    repack_kernel<<<448, 256, 0, stream>>>(c1w1, c1w2, projw, wcat, pwbf);

    // 3) conv1 implicit GEMM with WMMA (M=81920 / 16-row tiles)
    conv1_wmma_kernel<<<M1 / 16, 256, 0, stream>>>(x0, wcat, c1b1, c1b2, lens, x1);

    // 4) projection GEMM with WMMA (512 wave-tasks, 4 waves/block)
    proj_wmma_kernel<<<128, 128, 0, stream>>>(x1, pwbf, projb, emb0);

    // 5) encoder conv + mask + squash + layernorm
    enc_kernel<<<MP, 256, 0, stream>>>(emb0, ew1, eb1, ew2, eb2, lens, caps);

    // 6) caps layer 0: primary(windowed) -> 32 conv capsules, LN
    caps_route_kernel<32, false, 0><<<MP, 256, 0, stream>>>(caps, W0, B0, v0, nullptr);

    // 7) caps layer 1: conv(windowed) -> 31 class capsules (blank masked),
    //    lengths + output layernorm -> d_out
    caps_route_kernel<31, true, 1><<<MP, 256, 0, stream>>>(v0, W1, B1, nullptr, out);
}
